// Retention_41575283425482
// MI455X (gfx1250) — matrix operand order repaired and checked by reading, not compiled, not run on hardware
//
#include <hip/hip_runtime.h>
#include <hip/hip_bf16.h>
#include <math.h>
#include <stdint.h>

#define BB 4
#define SS 4096
#define DD 256
#define CC 128
#define NC 32              // SS/CC chunks per batch
#define GAMMA 0.9865f
#define CINV  0.0625f      // 1/sqrt(256)

typedef __bf16 bf16;
typedef __attribute__((ext_vector_type(16))) __bf16 v16bf;
typedef __attribute__((ext_vector_type(8)))  float  v8f;

// ---- scheduler group barriers (interleave WMMA with memory, CK-style) ----
#if defined(__has_builtin)
#if __has_builtin(__builtin_amdgcn_sched_group_barrier)
#define SCHED_GROUP(mask, size, id) __builtin_amdgcn_sched_group_barrier(mask, size, id)
#endif
#endif
#ifndef SCHED_GROUP
#define SCHED_GROUP(mask, size, id)
#endif
#define SG_WMMA    0x008  // MFMA/WMMA group
#define SG_VMEM_RD 0x020  // VMEM read group
#define SG_DS_RD   0x100  // LDS read group

// ---------------- WMMA fragment helpers (contiguous k = 16*hi + i, wave32) ----------------
static __device__ __forceinline__ v16bf load_a_frag(const bf16* p, int stride) {
  int lane = threadIdx.x & 31;
  int r = lane & 15, hi = lane >> 4;
  const bf16* row = p + (size_t)r * stride;
  v16bf f;
#pragma unroll
  for (int i = 0; i < 8; ++i) f[i] = row[16 * hi + i];
#pragma unroll
  for (int i = 0; i < 8; ++i) f[8 + i] = row[16 * hi + 8 + i];
  return f;
}

static __device__ __forceinline__ v16bf load_b_frag_nk(const bf16* p, int stride) {
  int lane = threadIdx.x & 31;
  int n = lane & 15, hi = lane >> 4;
  const bf16* row = p + (size_t)n * stride + 16 * hi;
  v16bf f;
#pragma unroll
  for (int i = 0; i < 16; ++i) f[i] = row[i];
  return f;
}

static __device__ __forceinline__ v8f wmma_bf16(v16bf a, v16bf b, v8f c) {
  return __builtin_amdgcn_wmma_f32_16x16x32_bf16(false, a, false, b, (short)0, c,
                                                 false, false);
}

// Async DMA: copy 16 contiguous bytes global -> LDS (ASYNCcnt-tracked).
static __device__ __forceinline__ void async_copy16(unsigned lds_off,
                                                    const void* gptr) {
  asm volatile("global_load_async_to_lds_b128 %0, %1, off"
               :: "v"(lds_off), "v"((unsigned long long)(uintptr_t)gptr)
               : "memory");
}
static __device__ __forceinline__ void wait_async0() {
  asm volatile("s_wait_asynccnt 0" ::: "memory");
}

// ---------------- decay tables ----------------
__global__ void k_tables(float* qsc, float* ksc) {
  int t = threadIdx.x;
  if (t < 128) qsc[t] = powf(GAMMA, (float)t);
  for (int j = t * 16; j < t * 16 + 16; ++j) {
    float r = (1.0f - powf(GAMMA, (float)(j + 1))) / (1.0f - GAMMA);
    ksc[j] = powf(GAMMA, -(float)(j & (CC - 1))) * rsqrtf(r);
  }
}

// W^T (bf16): WT[w][n][k] = W[k][n]
__global__ void k_wt(const float* Wq, const float* Wk, const float* Wv, bf16* WT) {
  int e = blockIdx.x * 256 + threadIdx.x;  // < 196608
  int w = e >> 16;
  int rem = e & 65535;
  int k = rem >> 8, n = rem & 255;
  const float* W = (w == 0) ? Wq : ((w == 1) ? Wk : Wv);
  WT[(w << 16) + n * 256 + k] = (bf16)W[rem];
}

// ---------------- projections ----------------
// Q/K written row-major (scaled); V written chunk-transposed VbT[b][c][d][q]
__global__ __launch_bounds__(256) void k_proj(
    const float* xq, const float* xk, const float* xv, const float* bq,
    const float* bk, const float* bv, const bf16* WT, const float* qsc,
    const float* ksc, bf16* Qb, bf16* Kb, bf16* VbT) {
  extern __shared__ char smem[];
  bf16* Xt = (bf16*)smem;  // [128][256]
  int mode = blockIdx.y;
  const float* X = (mode == 0) ? xq : ((mode == 1) ? xk : xv);
  const float* bias = (mode == 0) ? bq : ((mode == 1) ? bk : bv);
  const bf16* Wt = WT + (mode << 16);
  int r0 = blockIdx.x * CC;
  int t = threadIdx.x;
  for (int idx = t; idx < CC * DD; idx += 256)
    Xt[idx] = (bf16)X[(size_t)r0 * DD + idx];
  __syncthreads();

  int w = t >> 5, lane = t & 31, nl = lane & 15, hi = lane >> 4;
  v8f acc[16] = {};
  for (int ks = 0; ks < 8; ++ks) {
    v16bf a = load_a_frag(Xt + (16 * w) * DD + 32 * ks, DD);
    v16bf b0 = load_b_frag_nk(Wt + 0 * DD + 32 * ks, DD);
    v16bf b1 = load_b_frag_nk(Wt + 16 * DD + 32 * ks, DD);
#pragma unroll
    for (int nt = 0; nt < 16; ++nt) {
      v16bf bn = b0;
      if (nt + 2 < 16) bn = load_b_frag_nk(Wt + 16 * (nt + 2) * DD + 32 * ks, DD);
      acc[nt] = wmma_bf16(a, b0, acc[nt]);
      b0 = b1; b1 = bn;
    }
    // schedule: A from LDS, 3 B-frags ahead, then [wmma | 2 loads] steady state
    SCHED_GROUP(SG_DS_RD, 2, 0);
    SCHED_GROUP(SG_VMEM_RD, 6, 0);
#pragma unroll
    for (int i = 0; i < 13; ++i) {
      SCHED_GROUP(SG_WMMA, 1, 0);
      SCHED_GROUP(SG_VMEM_RD, 2, 0);
    }
    SCHED_GROUP(SG_WMMA, 3, 0);
  }
  float rs[8];
#pragma unroll
  for (int v = 0; v < 8; ++v) {
    int m = 16 * w + 8 * hi + v;
    rs[v] = (mode == 0) ? qsc[m]
            : (mode == 1) ? ksc[(r0 & (SS - 1)) + m]
                          : 1.0f;
  }
  int bidx = r0 >> 12;             // batch
  int cidx = (r0 & (SS - 1)) >> 7; // chunk
#pragma unroll
  for (int nt = 0; nt < 16; ++nt) {
    int n = 16 * nt + nl;
    float bb = bias[n];
#pragma unroll
    for (int v = 0; v < 8; ++v) {
      int m = 16 * w + 8 * hi + v;
      float val = (acc[nt][v] + bb) * rs[v];
      if (mode == 2) {
        VbT[(((size_t)(bidx * NC + cidx)) << 15) + (size_t)n * CC + m] = (bf16)val;
      } else {
        bf16* Out = (mode == 0) ? Qb : Kb;
        Out[(size_t)(r0 + m) * DD + n] = (bf16)val;
      }
    }
  }
}

// ---------------- denominator path ----------------
__global__ void k_dA(const bf16* Kb, float* dA, float g127) {
  int c = blockIdx.x, b = blockIdx.y, d = threadIdx.x;
  size_t row0 = (size_t)(b * SS + c * CC);
  float s = 0.f;
  for (int q = 0; q < CC; ++q) s += (float)Kb[(row0 + q) * DD + d];
  dA[((b * NC + c) << 8) + d] = g127 * s;
}

__global__ void k_Ascan(const float* dA, float* Aexc, float g128) {
  int b = blockIdx.x, d = threadIdx.x;
  float acc = 0.f;
  for (int c = 0; c < NC; ++c) {
    Aexc[((b * NC + c) << 8) + d] = acc;
    acc = g128 * acc + dA[((b * NC + c) << 8) + d];
  }
}

// rowsum[j] = cInv * Q_j . A_j ;  invd = 1/max(|rowsum|,1)
__global__ __launch_bounds__(256) void k_rowsum(const bf16* Qb, const bf16* Kb,
                                                const float* Aexc, float* invd) {
  __shared__ float sm[CC];
  int c = blockIdx.x, b = blockIdx.y, d = threadIdx.x;
  if (d < CC) sm[d] = 0.f;
  __syncthreads();
  size_t row0 = (size_t)(b * SS + c * CC);
  float a = Aexc[((b * NC + c) << 8) + d];
  float gp = 1.f, gip = 1.f;
  const float gi = 1.0f / GAMMA;
  for (int p = 0; p < CC; ++p) {
    float kv = (float)Kb[(row0 + p) * DD + d];
    float qv = (float)Qb[(row0 + p) * DD + d];
    a = GAMMA * a + gp * kv;
    float part = (qv * gip) * a;
#pragma unroll
    for (int off = 16; off >= 1; off >>= 1) part += __shfl_xor(part, off, 32);
    if ((d & 31) == 0) atomicAdd(&sm[p], part);
    gp *= GAMMA;
    gip *= gi;
  }
  __syncthreads();
  if (d < CC) {
    float den = fmaxf(fabsf(CINV * sm[d]), 1.0f);
    invd[b * SS + c * CC + d] = 1.0f / den;
  }
}

// ---------------- per-chunk state delta: dS = g127*cInv * (Kb/den)^T @ V ----------------
__global__ __launch_bounds__(256) void k_dS(const bf16* Kb, const bf16* VbT,
                                            const float* invd, float* dS,
                                            float g127) {
  extern __shared__ char smem[];
  bf16* At = (bf16*)smem;                  // [128][128]
  bf16* VT = (bf16*)(smem + CC * CC * 2);  // [256][128]
  int mh = blockIdx.x, c = blockIdx.y, b = blockIdx.z;
  size_t row0 = (size_t)(b * SS + c * CC);
  int t = threadIdx.x;
  {  // async DMA V^T tile (pure copy: VbT is already chunk-transposed)
    const bf16* src = VbT + (((size_t)(b * NC + c)) << 15);
    unsigned vt0 = (unsigned)(uintptr_t)VT;
    for (int i = t; i < 4096; i += 256) async_copy16(vt0 + i * 16, src + (size_t)i * 8);
  }
  for (int idx = t; idx < CC * CC; idx += 256) {
    int q = idx >> 7, d1l = idx & 127;
    float sc = g127 * CINV * invd[b * SS + c * CC + q];
    At[d1l * CC + q] = (bf16)((float)Kb[(row0 + q) * DD + 128 * mh + d1l] * sc);
  }
  wait_async0();
  __syncthreads();
  int w = t >> 5, lane = t & 31, nl = lane & 15, hi = lane >> 4;
  v8f acc[16] = {};
  for (int ks = 0; ks < 4; ++ks) {
    v16bf a = load_a_frag(At + (16 * w) * CC + 32 * ks, CC);
    v16bf b0 = load_b_frag_nk(VT + 0 * CC + 32 * ks, CC);
    v16bf b1 = load_b_frag_nk(VT + 16 * CC + 32 * ks, CC);
#pragma unroll
    for (int nt = 0; nt < 16; ++nt) {
      v16bf bn = b0;
      if (nt + 2 < 16) bn = load_b_frag_nk(VT + 16 * (nt + 2) * CC + 32 * ks, CC);
      acc[nt] = wmma_bf16(a, b0, acc[nt]);
      b0 = b1; b1 = bn;
    }
    SCHED_GROUP(SG_DS_RD, 8, 0);
#pragma unroll
    for (int i = 0; i < 13; ++i) {
      SCHED_GROUP(SG_WMMA, 1, 0);
      SCHED_GROUP(SG_DS_RD, 2, 0);
    }
    SCHED_GROUP(SG_WMMA, 3, 0);
  }
  size_t base = ((size_t)(b * NC + c)) << 16;
#pragma unroll
  for (int nt = 0; nt < 16; ++nt) {
    int n = 16 * nt + nl;
#pragma unroll
    for (int v = 0; v < 8; ++v) {
      int d1 = 128 * mh + 16 * w + 8 * hi + v;
      dS[base + (size_t)d1 * DD + n] = acc[nt][v];
    }
  }
}

// exclusive prefix scan of S; writes gamma*Sexc^T bf16 [B][NC][d2][d1]
__global__ void k_Sscan(const float* dS, bf16* ST, float g128) {
  int gid = blockIdx.x * 256 + threadIdx.x;
  int b = gid >> 16;
  int rem = gid & 65535;
  int d1 = rem >> 8, d2 = rem & 255;
  float acc = 0.f;
  for (int c = 0; c < NC; ++c) {
    size_t base = ((size_t)(b * NC + c)) << 16;
    ST[base + (size_t)d2 * DD + d1] = (bf16)(GAMMA * acc);
    acc = g128 * acc + dS[base + (size_t)d1 * DD + d2];
  }
}

// ---------------- output ----------------
__global__ __launch_bounds__(256) void k_out(const bf16* Qb, const bf16* Kb,
                                             const bf16* VbT, const bf16* ST,
                                             const float* invd, float* out) {
  extern __shared__ char smem[];
  bf16* Pt = (bf16*)smem;                  // [128][128]
  bf16* VT = (bf16*)(smem + CC * CC * 2);  // [256][128]
  int c = blockIdx.x, b = blockIdx.y;
  size_t row0 = (size_t)(b * SS + c * CC);
  int t = threadIdx.x;
  int w = t >> 5, lane = t & 31, nl = lane & 15, hi = lane >> 4;

  {  // async DMA V^T tile; overlaps the QK^T WMMA block below
    const bf16* src = VbT + (((size_t)(b * NC + c)) << 15);
    unsigned vt0 = (unsigned)(uintptr_t)VT;
    for (int i = t; i < 4096; i += 256) async_copy16(vt0 + i * 16, src + (size_t)i * 8);
  }
  const bf16* Sb = ST + (((size_t)(b * NC + c)) << 16);
  for (int i = t; i < 1024; i += 256)  // warm 128KB state tile into cache
    __builtin_prefetch((const char*)Sb + (size_t)i * 128, 0, 3);

  {  // P = Qtilde @ Ktilde^T
    v8f pacc[8] = {};
    for (int ks = 0; ks < 8; ++ks) {
      v16bf a = load_a_frag(Qb + (row0 + 16 * w) * DD + 32 * ks, DD);
      v16bf b0 = load_b_frag_nk(Kb + (row0 + 0) * DD + 32 * ks, DD);
      v16bf b1 = load_b_frag_nk(Kb + (row0 + 16) * DD + 32 * ks, DD);
#pragma unroll
      for (int nt = 0; nt < 8; ++nt) {
        v16bf bn = b0;
        if (nt + 2 < 8)
          bn = load_b_frag_nk(Kb + (row0 + 16 * (nt + 2)) * DD + 32 * ks, DD);
        pacc[nt] = wmma_bf16(a, b0, pacc[nt]);
        b0 = b1; b1 = bn;
      }
      SCHED_GROUP(SG_VMEM_RD, 8, 0);
#pragma unroll
      for (int i = 0; i < 5; ++i) {
        SCHED_GROUP(SG_WMMA, 1, 0);
        SCHED_GROUP(SG_VMEM_RD, 2, 0);
      }
      SCHED_GROUP(SG_WMMA, 3, 0);
    }
#pragma unroll
    for (int nt = 0; nt < 8; ++nt) {
      int q = 16 * nt + nl;
      float cs = CINV * invd[b * SS + c * CC + q];
#pragma unroll
      for (int v = 0; v < 8; ++v) {
        int p = 16 * w + 8 * hi + v;
        float val = (p >= q) ? pacc[nt][v] * cs : 0.0f;
        Pt[p * CC + q] = (bf16)val;
      }
    }
  }
  wait_async0();
  __syncthreads();

  v8f acc[16] = {};
  for (int ks = 0; ks < 4; ++ks) {  // intra: P @ V
    v16bf a = load_a_frag(Pt + (16 * w) * CC + 32 * ks, CC);
    v16bf b0 = load_b_frag_nk(VT + 0 * CC + 32 * ks, CC);
    v16bf b1 = load_b_frag_nk(VT + 16 * CC + 32 * ks, CC);
#pragma unroll
    for (int nt = 0; nt < 16; ++nt) {
      v16bf bn = b0;
      if (nt + 2 < 16) bn = load_b_frag_nk(VT + 16 * (nt + 2) * CC + 32 * ks, CC);
      acc[nt] = wmma_bf16(a, b0, acc[nt]);
      b0 = b1; b1 = bn;
    }
    SCHED_GROUP(SG_DS_RD, 8, 0);
#pragma unroll
    for (int i = 0; i < 13; ++i) {
      SCHED_GROUP(SG_WMMA, 1, 0);
      SCHED_GROUP(SG_DS_RD, 2, 0);
    }
    SCHED_GROUP(SG_WMMA, 3, 0);
  }
  for (int ks = 0; ks < 8; ++ks) {  // cross: Qtilde @ (gamma*Sexc)
    v16bf a = load_a_frag(Qb + (row0 + 16 * w) * DD + 32 * ks, DD);
    v16bf b0 = load_b_frag_nk(Sb + 0 * DD + 32 * ks, DD);
    v16bf b1 = load_b_frag_nk(Sb + 16 * DD + 32 * ks, DD);
#pragma unroll
    for (int nt = 0; nt < 16; ++nt) {
      v16bf bn = b0;
      if (nt + 2 < 16) bn = load_b_frag_nk(Sb + 16 * (nt + 2) * DD + 32 * ks, DD);
      acc[nt] = wmma_bf16(a, b0, acc[nt]);
      b0 = b1; b1 = bn;
    }
    SCHED_GROUP(SG_VMEM_RD, 8, 0);
#pragma unroll
    for (int i = 0; i < 13; ++i) {
      SCHED_GROUP(SG_WMMA, 1, 0);
      SCHED_GROUP(SG_VMEM_RD, 2, 0);
    }
    SCHED_GROUP(SG_WMMA, 3, 0);
  }
#pragma unroll
  for (int nt = 0; nt < 16; ++nt) {
    int n = 16 * nt + nl;
#pragma unroll
    for (int v = 0; v < 8; ++v) {
      int p = 16 * w + 8 * hi + v;
      out[(row0 + p) * DD + n] = acc[nt][v];
    }
  }
}

extern "C" void kernel_launch(void* const* d_in, const int* in_sizes, int n_in,
                              void* d_out, int out_size, void* d_ws,
                              size_t ws_size, hipStream_t stream) {
  (void)in_sizes; (void)n_in; (void)out_size; (void)ws_size;
  const float* xq = (const float*)d_in[0];
  const float* xk = (const float*)d_in[1];
  const float* xv = (const float*)d_in[2];
  const float* Wq = (const float*)d_in[3];
  const float* bq = (const float*)d_in[4];
  const float* Wk = (const float*)d_in[5];
  const float* bk = (const float*)d_in[6];
  const float* Wv = (const float*)d_in[7];
  const float* bv = (const float*)d_in[8];
  float* out = (float*)d_out;

  char* ws = (char*)d_ws;
  size_t off = 0;
  auto alloc = [&](size_t bytes) -> void* {
    void* p = ws + off;
    off = (off + bytes + 255) & ~(size_t)255;
    return p;
  };
  bf16* WT   = (bf16*)alloc((size_t)3 * 65536 * 2);
  float* qsc = (float*)alloc(128 * 4);
  float* ksc = (float*)alloc(4096 * 4);
  bf16* Qb   = (bf16*)alloc((size_t)BB * SS * DD * 2);
  bf16* Kb   = (bf16*)alloc((size_t)BB * SS * DD * 2);
  bf16* VbT  = (bf16*)alloc((size_t)BB * NC * 32768 * 2);
  float* dA  = (float*)alloc((size_t)BB * NC * DD * 4);
  float* Aex = (float*)alloc((size_t)BB * NC * DD * 4);
  float* ivd = (float*)alloc((size_t)BB * SS * 4);
  float* dS  = (float*)alloc((size_t)BB * NC * 65536 * 4);
  bf16* ST   = (bf16*)alloc((size_t)BB * NC * 65536 * 2);

  float g127 = (float)pow((double)GAMMA, 127.0);
  float g128 = (float)pow((double)GAMMA, 128.0);

  k_tables<<<1, 256, 0, stream>>>(qsc, ksc);
  k_wt<<<768, 256, 0, stream>>>(Wq, Wk, Wv, WT);
  k_proj<<<dim3(128, 3), 256, CC * DD * 2, stream>>>(xq, xk, xv, bq, bk, bv, WT,
                                                     qsc, ksc, Qb, Kb, VbT);
  k_dA<<<dim3(NC, BB), 256, 0, stream>>>(Kb, dA, g127);
  k_Ascan<<<BB, 256, 0, stream>>>(dA, Aex, g128);
  k_rowsum<<<dim3(NC, BB), 256, 0, stream>>>(Qb, Kb, Aex, ivd);
  k_dS<<<dim3(2, NC, BB), 256, CC * CC * 2 + CC * DD * 2, stream>>>(Kb, VbT, ivd,
                                                                    dS, g127);
  k_Sscan<<<1024, 256, 0, stream>>>(dS, ST, g128);
  k_out<<<dim3(NC, BB), 256, CC * CC * 2 + CC * DD * 2, stream>>>(Qb, Kb, VbT,
                                                                  ST, ivd, out);
}
